// MoE_377957122269
// MI455X (gfx1250) — matrix operand order repaired and checked by reading, not compiled, not run on hardware
//
#include <hip/hip_runtime.h>

// ---------------------------------------------------------------------------
// MoE (sigmoid-gated top-2, 8 routed experts + 1 shared) for MI455X / gfx1250.
//   h'[n, e*H+j] = C[n,e] * silu(x·w1[e]T) * (x·w3[e]T)   (zero if unselected)
//   out[n, d]   = sum_k h'[n,k] * w2t[d,k]                (K = 9*H = 9216)
// GEMM1 is token-compacted per expert (top-2 => 4x less routed work than dense)
// with deterministic prefix-sum compaction (no atomics). GEMMs run on
// v_wmma_f32_16x16x32_bf16 with TDM (tensor_load_to_lds) staged LDS tiles.
// ---------------------------------------------------------------------------

#define D_DIM 2048
#define H_DIM 1024
#define N_TOK 2048
#define NEXP  8
#define SLOTS 9
#define K2    (SLOTS * H_DIM)   // 9216

#if defined(__gfx1250__) && __has_builtin(__builtin_amdgcn_tensor_load_to_lds) && \
    __has_builtin(__builtin_amdgcn_s_wait_tensorcnt)
#define USE_TDM 1
#endif

typedef __attribute__((ext_vector_type(16))) __bf16 v16bf;
typedef __attribute__((ext_vector_type(8)))  __bf16 v8bf;
typedef __attribute__((ext_vector_type(8)))  float  v8f;
typedef __attribute__((ext_vector_type(4)))  unsigned int u32x4;
typedef __attribute__((ext_vector_type(8)))  int i32x8;
typedef __attribute__((ext_vector_type(4)))  int i32x4;

__device__ __forceinline__ unsigned short f2bf(float f) {
    unsigned int u = __float_as_uint(f);
    unsigned int r = u + 0x7FFFu + ((u >> 16) & 1u);   // round-to-nearest-even
    return (unsigned short)(r >> 16);
}

// ---------------------------------------------------------------------------
// TDM: 2D tile load Global->LDS (CDNA5 ISA).
// data_size=2B; LDS pad: every 16 DWORDs (64B row) insert 4 DWORDs (16B)
// -> LDS row pitch 80B == 40 halves, matching the fragment loaders.
// ---------------------------------------------------------------------------
__device__ __forceinline__ void tdm_load_2d(unsigned lds_addr, const void* gptr,
                                            unsigned tile_rows, unsigned tile_elems,
                                            unsigned long long row_stride_elems,
                                            unsigned tensor_d0, unsigned tensor_d1)
{
#if defined(USE_TDM)
    unsigned long long ga = (unsigned long long)gptr;
    u32x4 g0;
    g0[0] = 1u;                                                 // count=1 (valid)
    g0[1] = lds_addr;                                           // LDS byte address
    g0[2] = (unsigned)(ga & 0xFFFFFFFFu);                       // global_addr lo
    g0[3] = (unsigned)((ga >> 32) & 0x1FFFFFFu) | (2u << 30);   // addr hi | type=2
    i32x8 g1;
    const unsigned w0 = (1u << 16)    // data_size = 2 bytes
                      | (1u << 20)    // pad_enable
                      | (3u << 22)    // pad_interval: 16 DWORDs (64B)
                      | (3u << 25);   // pad_amount:   4 DWORDs (16B)
    g1[0] = (int)w0;
    g1[1] = (int)((tensor_d0 & 0xFFFFu) << 16);
    g1[2] = (int)(((tensor_d0 >> 16) & 0xFFFFu) | ((tensor_d1 & 0xFFFFu) << 16));
    g1[3] = (int)(((tensor_d1 >> 16) & 0xFFFFu) | ((tile_elems & 0xFFFFu) << 16));
    g1[4] = (int)(tile_rows & 0xFFFFu);                         // tile_dim1, tile_dim2=0
    g1[5] = (int)(unsigned)(row_stride_elems & 0xFFFFFFFFull);  // dim0_stride lo32
    g1[6] = (int)(unsigned)((row_stride_elems >> 32) & 0xFFFFull);
    g1[7] = 0;
    i32x4 gz4 = {0, 0, 0, 0};
    i32x8 gz8 = {0, 0, 0, 0, 0, 0, 0, 0};
    __builtin_amdgcn_tensor_load_to_lds(g0, g1, gz4, gz4, gz8, 0);
#else
    (void)lds_addr; (void)gptr; (void)tile_rows; (void)tile_elems;
    (void)row_stride_elems; (void)tensor_d0; (void)tensor_d1;
#endif
}

// ---------------------------------------------------------------------------
// Router: one wave per token; sigmoid scores, biased top-2 (first-index ties),
// writes dense combine row Cw[n][0..8] (slot 8 = shared expert, weight 1.0).
// ---------------------------------------------------------------------------
__global__ __launch_bounds__(256) void router_kernel(
    const float* __restrict__ x, const float* __restrict__ gw,
    const float* __restrict__ ebias, float* __restrict__ Cw)
{
    const int wv   = threadIdx.x >> 5;
    const int lane = threadIdx.x & 31;
    const int n    = blockIdx.x * 8 + wv;
    if (n >= N_TOK) return;

    float xr[64];
#pragma unroll
    for (int i = 0; i < 64; ++i)
        xr[i] = x[(size_t)n * D_DIM + lane + 32 * i];

    float sc[NEXP];
    for (int e = 0; e < NEXP; ++e) {
        float acc = 0.f;
#pragma unroll
        for (int i = 0; i < 64; ++i)
            acc += xr[i] * gw[(size_t)e * D_DIM + lane + 32 * i];
#pragma unroll
        for (int off = 16; off > 0; off >>= 1)
            acc += __shfl_xor(acc, off, 32);
        sc[e] = 1.0f / (1.0f + __expf(-acc));
    }

    if (lane == 0) {
        float bb[NEXP];
        for (int e = 0; e < NEXP; ++e) bb[e] = sc[e] + ebias[e];
        int i0 = 0;
        for (int e = 1; e < NEXP; ++e) if (bb[e] > bb[i0]) i0 = e;
        int i1 = -1; float best = -1e30f;
        for (int e = 0; e < NEXP; ++e)
            if (e != i0 && bb[e] > best) { best = bb[e]; i1 = e; }
        float s = sc[i0] + sc[i1] + 1e-20f;
        float w[SLOTS];
        for (int j = 0; j < SLOTS; ++j) w[j] = 0.f;
        w[i0] = sc[i0] / s;
        w[i1] = sc[i1] / s;
        w[8]  = 1.0f;
        for (int j = 0; j < SLOTS; ++j) Cw[(size_t)n * SLOTS + j] = w[j];
    }
}

// ---------------------------------------------------------------------------
// Deterministic per-expert token compaction (single block, prefix sums).
// tokList[e][i] ascending; padded to a multiple of 64 with the last token
// (repeats are fine: pad rows are masked at the h' store).
// ---------------------------------------------------------------------------
__global__ __launch_bounds__(256) void compact_kernel(
    const float* __restrict__ Cw, int* __restrict__ tokList, int* __restrict__ tokCnt)
{
    __shared__ int scanBuf[256];
    __shared__ int baseSh;
    const int tid = threadIdx.x;

    for (int e = 0; e < NEXP; ++e) {
        if (tid == 0) baseSh = 0;
        __syncthreads();
        for (int c = 0; c < N_TOK / 256; ++c) {
            const int n = c * 256 + tid;
            const int flag = (Cw[(size_t)n * SLOTS + e] > 0.f) ? 1 : 0;
            scanBuf[tid] = flag;
            __syncthreads();
            for (int off = 1; off < 256; off <<= 1) {   // inclusive scan
                const int v = scanBuf[tid];
                const int u = (tid >= off) ? scanBuf[tid - off] : 0;
                __syncthreads();
                scanBuf[tid] = v + u;
                __syncthreads();
            }
            const int incl = scanBuf[tid];
            const int base = baseSh;
            if (flag) tokList[(size_t)e * N_TOK + base + incl - 1] = n;
            __syncthreads();
            if (tid == 255) baseSh = base + incl;
            __syncthreads();
        }
        const int cnt = baseSh;
        if (cnt > 0) {
            const int pad  = ((cnt + 63) / 64) * 64;
            const int last = tokList[(size_t)e * N_TOK + cnt - 1];
            for (int i = cnt + tid; i < pad; i += 256)
                tokList[(size_t)e * N_TOK + i] = last;
        }
        if (tid == 0) tokCnt[e] = cnt;
        __syncthreads();
    }
}

// ---------------------------------------------------------------------------
// fp32 -> bf16 conversions / weight re-layout / hp zero-fill
// ---------------------------------------------------------------------------
__global__ void cvt_x_kernel(const float* __restrict__ x,
                             unsigned short* __restrict__ xb, long long n)
{
    long long i = (long long)blockIdx.x * blockDim.x + threadIdx.x;
    long long stride = (long long)gridDim.x * blockDim.x;
    for (; i < n; i += stride) xb[i] = f2bf(x[i]);
}

__global__ void zero_hp_kernel(uint4* __restrict__ hp, long long n16)
{
    const uint4 z = {0u, 0u, 0u, 0u};
    long long i = (long long)blockIdx.x * blockDim.x + threadIdx.x;
    long long stride = (long long)gridDim.x * blockDim.x;
    for (; i < n16; i += stride) hp[i] = z;
}

__global__ void cvt_w13_kernel(const float* __restrict__ ew1,
                               const float* __restrict__ ew3,
                               const float* __restrict__ sw1,
                               const float* __restrict__ sw3,
                               unsigned short* __restrict__ w13)
{
    const long long HD = (long long)H_DIM * D_DIM;
    const long long total = (long long)SLOTS * 2 * HD;
    long long i = (long long)blockIdx.x * blockDim.x + threadIdx.x;
    long long stride = (long long)gridDim.x * blockDim.x;
    for (; i < total; i += stride) {
        long long s   = i / (2 * HD);
        long long rem = i % (2 * HD);
        long long p   = rem / HD;
        long long j   = rem % HD;
        float v;
        if (s < NEXP) v = p ? ew3[s * HD + j] : ew1[s * HD + j];
        else          v = p ? sw3[j]          : sw1[j];
        w13[i] = f2bf(v);
    }
}

__global__ void cvt_w2t_kernel(const float* __restrict__ ew2,
                               const float* __restrict__ sw2,
                               unsigned short* __restrict__ w2t)
{
    const long long DH = (long long)D_DIM * H_DIM;
    const long long total = (long long)SLOTS * DH;
    long long i = (long long)blockIdx.x * blockDim.x + threadIdx.x;
    long long stride = (long long)gridDim.x * blockDim.x;
    for (; i < total; i += stride) {
        long long e   = i / DH;
        long long rem = i % DH;
        long long d   = rem / H_DIM;
        long long h   = rem % H_DIM;
        float v = (e < NEXP) ? ew2[e * DH + d * H_DIM + h] : sw2[d * H_DIM + h];
        w2t[d * (long long)K2 + e * H_DIM + h] = f2bf(v);
    }
}

// ---------------------------------------------------------------------------
// Fragments (ISA 16-bit layouts, LDS pitch 40 halves / 80B).
// ---------------------------------------------------------------------------
__device__ __forceinline__ v16bf load_a_frag(const unsigned short* lA,
                                             int rowTile, int lane)
{
    const int m    = lane & 15;
    const int koff = (lane >> 4) * 8;
    const unsigned short* base = lA + (rowTile * 16 + m) * 40;
    v8bf lo = *(const v8bf*)(base + koff);
    v8bf hi = *(const v8bf*)(base + koff + 16);
    return __builtin_shufflevector(lo, hi, 0,1,2,3,4,5,6,7,8,9,10,11,12,13,14,15);
}

__device__ __forceinline__ v16bf load_b_frag(const unsigned short* lB,
                                             int colTile, int lane)
{
    const int c  = lane & 15;
    const int kb = (lane >> 4) * 8;
    const unsigned short* base = lB + (colTile * 16 + c) * 40 + kb;
    v8bf lo = *(const v8bf*)(base);
    v8bf hi = *(const v8bf*)(base + 16);
    return __builtin_shufflevector(lo, hi, 0,1,2,3,4,5,6,7,8,9,10,11,12,13,14,15);
}

__device__ __forceinline__ v8f wmma_bf16(v16bf a, v16bf b, v8f c) {
    return __builtin_amdgcn_wmma_f32_16x16x32_bf16(false, a, false, b,
                                                   (short)0, c, false, false);
}

// ---------------------------------------------------------------------------
// GEMM1 (token-compacted): for slot e, rows are gathered tokens from tokList.
// Block 256 thr = 8 waves (2x4 of 32x32) -> 64 tokens x 128 H columns.
// A (gathered rows): per-thread b128 loads w/ register prefetch.
// B (w1,w3 tiles, contiguous): TDM double-buffered, s_wait_tensorcnt pipeline.
// ---------------------------------------------------------------------------
__global__ __launch_bounds__(256) void gemm1_kernel(
    const unsigned short* __restrict__ xb,
    const unsigned short* __restrict__ w13,
    const float* __restrict__ Cw,
    const int* __restrict__ tokList,
    const int* __restrict__ tokCnt,
    unsigned short* __restrict__ hp)
{
    const int slot  = blockIdx.z;
    const int row0c = blockIdx.y * 64;     // position in expert's (padded) list
    const int col0  = blockIdx.x * 128;    // H tile

    int cnt;
    if (slot < NEXP) {
        cnt = tokCnt[slot];
        if (row0c >= cnt) return;          // padded-grid early exit
    } else {
        cnt = N_TOK;                       // shared expert: identity list
    }

    const unsigned short* w1 = w13 + (size_t)slot * 2 * H_DIM * D_DIM;
    const unsigned short* w3 = w1 + (size_t)H_DIM * D_DIM;

    __shared__ __align__(16) unsigned short lA [64 * 40];
    __shared__ __align__(16) unsigned short lB1[2][128 * 40];
    __shared__ __align__(16) unsigned short lB3[2][128 * 40];
    __shared__ int   ldsTok[64];
    __shared__ float sScale[64];

    const int tid = threadIdx.x;
    if (tid < 64) {
        int t;
        if (slot < NEXP) {
            int idx = row0c + tid;
            if (idx >= cnt) idx = cnt - 1;             // clamp; store is masked
            t = tokList[(size_t)slot * N_TOK + idx];
        } else {
            t = row0c + tid;
        }
        ldsTok[tid] = t;
        sScale[tid] = (slot < NEXP) ? Cw[(size_t)t * SLOTS + slot] : 1.0f;
    }
    __syncthreads();

    const int lane = tid & 31;
    const int wv   = tid >> 5;
    const int wr   = wv >> 2;        // 0..1 : 32-row group
    const int wcq  = wv & 3;         // 0..3 : 32-col group

    v8f acc1[2][2] = {{{}, {}}, {{}, {}}};
    v8f acc3[2][2] = {{{}, {}}, {{}, {}}};

    const int srow = tid >> 2;             // 0..63
    const int soff = (tid & 3) * 8;
    const size_t aRowBase = (size_t)ldsTok[srow] * D_DIM;   // gathered A row

#if defined(USE_TDM)
    int bufB = 0;
    if (tid < 32) {
        tdm_load_2d((unsigned)(size_t)&lB1[0][0], w1 + (size_t)col0 * D_DIM,
                    128, 32, D_DIM, D_DIM, H_DIM);
        tdm_load_2d((unsigned)(size_t)&lB3[0][0], w3 + (size_t)col0 * D_DIM,
                    128, 32, D_DIM, D_DIM, H_DIM);
    }
    uint4 pa = *(const uint4*)&xb[aRowBase + soff];
    for (int k0 = 0; k0 < D_DIM; k0 += 32) {
        __syncthreads();                   // prev compute done; lA free
        *(uint4*)&lA[srow * 40 + soff] = pa;
        const int kn = k0 + 32;
        if (kn < D_DIM)
            pa = *(const uint4*)&xb[aRowBase + kn + soff];
        if (tid < 32) {
            if (kn < D_DIM) {
                tdm_load_2d((unsigned)(size_t)&lB1[bufB ^ 1][0],
                            w1 + (size_t)col0 * D_DIM + kn, 128, 32, D_DIM, D_DIM, H_DIM);
                tdm_load_2d((unsigned)(size_t)&lB3[bufB ^ 1][0],
                            w3 + (size_t)col0 * D_DIM + kn, 128, 32, D_DIM, D_DIM, H_DIM);
                __builtin_amdgcn_s_wait_tensorcnt(2);   // current B tiles done
            } else {
                __builtin_amdgcn_s_wait_tensorcnt(0);   // drain
            }
        }
        __syncthreads();                   // lA stored + current B ready

        v16bf aF[2];
#pragma unroll
        for (int ri = 0; ri < 2; ++ri)
            aF[ri] = load_a_frag(lA, wr * 2 + ri, lane);
#pragma unroll
        for (int ci = 0; ci < 2; ++ci) {
            v16bf b1F = load_b_frag(&lB1[bufB][0], wcq * 2 + ci, lane);
            v16bf b3F = load_b_frag(&lB3[bufB][0], wcq * 2 + ci, lane);
#pragma unroll
            for (int ri = 0; ri < 2; ++ri) {
                acc1[ri][ci] = wmma_bf16(aF[ri], b1F, acc1[ri][ci]);
                acc3[ri][ci] = wmma_bf16(aF[ri], b3F, acc3[ri][ci]);
            }
        }
        bufB ^= 1;
    }
#else
    for (int k0 = 0; k0 < D_DIM; k0 += 32) {
        __syncthreads();
        *(uint4*)&lA[srow * 40 + soff] =
            *(const uint4*)&xb[aRowBase + k0 + soff];
#pragma unroll
        for (int j = 0; j < 2; ++j) {
            *(uint4*)&lB1[0][(srow + 64 * j) * 40 + soff] =
                *(const uint4*)&w1[(size_t)(col0 + srow + 64 * j) * D_DIM + k0 + soff];
            *(uint4*)&lB3[0][(srow + 64 * j) * 40 + soff] =
                *(const uint4*)&w3[(size_t)(col0 + srow + 64 * j) * D_DIM + k0 + soff];
        }
        __syncthreads();
        v16bf aF[2];
#pragma unroll
        for (int ri = 0; ri < 2; ++ri)
            aF[ri] = load_a_frag(lA, wr * 2 + ri, lane);
#pragma unroll
        for (int ci = 0; ci < 2; ++ci) {
            v16bf b1F = load_b_frag(&lB1[0][0], wcq * 2 + ci, lane);
            v16bf b3F = load_b_frag(&lB3[0][0], wcq * 2 + ci, lane);
#pragma unroll
            for (int ri = 0; ri < 2; ++ri) {
                acc1[ri][ci] = wmma_bf16(aF[ri], b1F, acc1[ri][ci]);
                acc3[ri][ci] = wmma_bf16(aF[ri], b3F, acc3[ri][ci]);
            }
        }
    }
#endif

    // epilogue: fused SwiGLU + routing scale; pad rows masked out
    const bool sharedSlot = (slot == NEXP);
#pragma unroll
    for (int ri = 0; ri < 2; ++ri) {
#pragma unroll
        for (int ci = 0; ci < 2; ++ci) {
#pragma unroll
            for (int i = 0; i < 8; ++i) {
                const int r  = wr * 32 + ri * 16 + i + ((lane >> 4) << 3);
                const int cc = wcq * 32 + ci * 16 + (lane & 15);
                if (sharedSlot || (row0c + r) < cnt) {
                    const int t = ldsTok[r];
                    const float a1 = acc1[ri][ci][i];
                    const float a3 = acc3[ri][ci][i];
                    const float sig = 1.0f / (1.0f + __expf(-a1));
                    const float hv  = sScale[r] * a1 * sig * a3;
                    hp[(size_t)t * K2 + (size_t)slot * H_DIM + col0 + cc] = f2bf(hv);
                }
            }
        }
    }
}

// ---------------------------------------------------------------------------
// GEMM2: out[n,d] = sum_{k<9216} h'[n,k] * w2t[d,k]
// TDM double-buffered staging: 2 tensor loads per stage (A, w2t).
// ---------------------------------------------------------------------------
__global__ __launch_bounds__(256) void gemm2_kernel(
    const unsigned short* __restrict__ hp,
    const unsigned short* __restrict__ w2t,
    float* __restrict__ out)
{
    const int row0 = blockIdx.y * 64;      // token tile
    const int col0 = blockIdx.x * 128;     // D tile

    __shared__ __align__(16) unsigned short lA[2][64  * 40];
    __shared__ __align__(16) unsigned short lB[2][128 * 40];

    const int tid  = threadIdx.x;
    const int lane = tid & 31;
    const int wv   = tid >> 5;
    const int wr   = wv >> 2;
    const int wcq  = wv & 3;

    v8f acc[2][2] = {{{}, {}}, {{}, {}}};

#if defined(USE_TDM)
    int buf = 0;
    if (tid < 32) {
        tdm_load_2d((unsigned)(size_t)&lA[0][0],
                    hp + (size_t)row0 * K2, 64, 32, K2, K2, N_TOK);
        tdm_load_2d((unsigned)(size_t)&lB[0][0],
                    w2t + (size_t)col0 * K2, 128, 32, K2, K2, D_DIM);
    }
    for (int k0 = 0; k0 < K2; k0 += 32) {
        if (tid < 32) {
            const int kn = k0 + 32;
            if (kn < K2) {
                tdm_load_2d((unsigned)(size_t)&lA[buf ^ 1][0],
                            hp + (size_t)row0 * K2 + kn, 64, 32, K2, K2, N_TOK);
                tdm_load_2d((unsigned)(size_t)&lB[buf ^ 1][0],
                            w2t + (size_t)col0 * K2 + kn, 128, 32, K2, K2, D_DIM);
                __builtin_amdgcn_s_wait_tensorcnt(2);   // current buffer done
            } else {
                __builtin_amdgcn_s_wait_tensorcnt(0);   // drain
            }
        }
        __syncthreads();

        v16bf aF[2];
#pragma unroll
        for (int ri = 0; ri < 2; ++ri)
            aF[ri] = load_a_frag(&lA[buf][0], wr * 2 + ri, lane);
#pragma unroll
        for (int ci = 0; ci < 2; ++ci) {
            v16bf bF = load_b_frag(&lB[buf][0], wcq * 2 + ci, lane);
#pragma unroll
            for (int ri = 0; ri < 2; ++ri)
                acc[ri][ci] = wmma_bf16(aF[ri], bF, acc[ri][ci]);
        }
        __syncthreads();
        buf ^= 1;
    }
#else
    const int srow = tid >> 2;
    const int soff = (tid & 3) * 8;
    for (int k0 = 0; k0 < K2; k0 += 32) {
        __syncthreads();
        *(uint4*)&lA[0][srow * 40 + soff] =
            *(const uint4*)&hp[(size_t)(row0 + srow) * K2 + k0 + soff];
#pragma unroll
        for (int j = 0; j < 2; ++j)
            *(uint4*)&lB[0][(srow + 64 * j) * 40 + soff] =
                *(const uint4*)&w2t[(size_t)(col0 + srow + 64 * j) * K2 + k0 + soff];
        __syncthreads();
        v16bf aF[2];
#pragma unroll
        for (int ri = 0; ri < 2; ++ri)
            aF[ri] = load_a_frag(&lA[0][0], wr * 2 + ri, lane);
#pragma unroll
        for (int ci = 0; ci < 2; ++ci) {
            v16bf bF = load_b_frag(&lB[0][0], wcq * 2 + ci, lane);
#pragma unroll
            for (int ri = 0; ri < 2; ++ri)
                acc[ri][ci] = wmma_bf16(aF[ri], bF, acc[ri][ci]);
        }
    }
#endif

#pragma unroll
    for (int ri = 0; ri < 2; ++ri) {
#pragma unroll
        for (int ci = 0; ci < 2; ++ci) {
#pragma unroll
            for (int i = 0; i < 8; ++i) {
                const int r  = wr * 32 + ri * 16 + i + ((lane >> 4) << 3);
                const int cc = wcq * 32 + ci * 16 + (lane & 15);
                out[(size_t)(row0 + r) * D_DIM + col0 + cc] = acc[ri][ci][i];
            }
        }
    }
}

// ---------------------------------------------------------------------------
// Launcher
// ---------------------------------------------------------------------------
extern "C" void kernel_launch(void* const* d_in, const int* in_sizes, int n_in,
                              void* d_out, int out_size, void* d_ws, size_t ws_size,
                              hipStream_t stream)
{
    const float* x      = (const float*)d_in[0];
    const float* gate_w = (const float*)d_in[1];
    const float* ew1    = (const float*)d_in[2];
    const float* ew2    = (const float*)d_in[3];
    const float* ew3    = (const float*)d_in[4];
    const float* sw1    = (const float*)d_in[5];
    const float* sw2    = (const float*)d_in[6];
    const float* sw3    = (const float*)d_in[7];
    const float* ebias  = (const float*)d_in[8];
    float* out          = (float*)d_out;

    char* ws = (char*)d_ws;
    unsigned short* xb  = (unsigned short*)ws; ws += (size_t)N_TOK * D_DIM * 2;
    unsigned short* w13 = (unsigned short*)ws; ws += (size_t)SLOTS * 2 * H_DIM * D_DIM * 2;
    unsigned short* w2t = (unsigned short*)ws; ws += (size_t)D_DIM * K2 * 2;
    unsigned short* hp  = (unsigned short*)ws; ws += (size_t)N_TOK * K2 * 2;
    float* Cw           = (float*)ws;          ws += (size_t)N_TOK * SLOTS * 4;
    int* tokList        = (int*)ws;            ws += (size_t)NEXP * N_TOK * 4;
    int* tokCnt         = (int*)ws;

    cvt_x_kernel<<<2048, 256, 0, stream>>>(x, xb, (long long)N_TOK * D_DIM);
    cvt_w13_kernel<<<4096, 256, 0, stream>>>(ew1, ew3, sw1, sw3, w13);
    cvt_w2t_kernel<<<4096, 256, 0, stream>>>(ew2, sw2, w2t);
    zero_hp_kernel<<<4096, 256, 0, stream>>>((uint4*)hp,
                                             (long long)N_TOK * K2 * 2 / 16);
    router_kernel<<<N_TOK / 8, 256, 0, stream>>>(x, gate_w, ebias, Cw);
    compact_kernel<<<1, 256, 0, stream>>>(Cw, tokList, tokCnt);

    gemm1_kernel<<<dim3(H_DIM / 128, N_TOK / 64, SLOTS), 256, 0, stream>>>(
        xb, w13, Cw, tokList, tokCnt, hp);
    gemm2_kernel<<<dim3(D_DIM / 128, N_TOK / 64, 1), 256, 0, stream>>>(hp, w2t, out);
}
